// Head_33552284516929
// MI455X (gfx1250) — matrix operand order repaired and checked by reading, not compiled, not run on hardware
//
#include <hip/hip_runtime.h>

typedef __attribute__((ext_vector_type(16))) _Float16 v16h;
typedef __attribute__((ext_vector_type(8)))  _Float16 v8h;
typedef __attribute__((ext_vector_type(8)))  float    v8f;
typedef __attribute__((ext_vector_type(4)))  float    v4f;

#define WMMA16x16x32(A, B, C) \
    __builtin_amdgcn_wmma_f32_16x16x32_f16(false, (A), false, (B), (short)0, (C), false, false)

static constexpr int T_SEQ = 256;       // sequence length
static constexpr int CIN   = 768;       // channels
static constexpr int HDIM  = 64;        // head size
static constexpr int BT    = 256 * 256; // total rows

// ---------------------------------------------------------------------------
// Kernel 0: convert Wq/Wk/Wv (fp32, [768,64]) into f16 tiles pre-swizzled into
// this code's contiguous-K B-matrix per-lane order (not AMD's table order;
// lanes 0-15 hold K=0..15 contiguously, lanes 16-31 hold K=16..31):
//   wswz[(((w*24 + kc)*4 + nt)*32 + lane)*16 + i]
//   K  = kc*32 + (lane<16 ? 0 : 16) + i,  column n = nt*16 + (lane&15)
// so the GEMM hot loop loads each B tile as one contiguous 32B chunk per lane.
// ---------------------------------------------------------------------------
__global__ __launch_bounds__(256) void swizzle_weights(
    const float* __restrict__ Wq, const float* __restrict__ Wk,
    const float* __restrict__ Wv, _Float16* __restrict__ wswz)
{
    int idx = blockIdx.x * 256 + threadIdx.x;
    if (idx >= 3 * 24 * 4 * 32 * 16) return;
    int i    = idx & 15;
    int lane = (idx >> 4) & 31;
    int nt   = (idx >> 9) & 3;
    int rem  = idx >> 11;       // w*24 + kc
    int kc   = rem % 24;
    int w    = rem / 24;
    int k    = kc * 32 + ((lane & 16) ? 16 : 0) + i;
    int n    = nt * 16 + (lane & 15);
    const float* W = (w == 0) ? Wq : (w == 1) ? Wk : Wv;
    wswz[idx] = (_Float16)W[k * HDIM + n];
}

// helper: B tile flat index -> pointer (flat = kc*12 + t, t = w*4+nt)
__device__ __forceinline__ const v16h* btile(const _Float16* wswz, int flat, int lane)
{
    flat %= 24 * 12;
    int kc = flat / 12;
    int t  = flat - kc * 12;
    int w  = t >> 2;
    int nt = t & 3;
    return (const v16h*)(wswz + (size_t)(((w * 24 + kc) * 4 + nt) * 512) + lane * 16);
}

// ---------------------------------------------------------------------------
// Kernel 1: fused QKV projection (WMMA f16->f32) + RoPE epilogue.
// Block = 256 threads = 8 waves; each wave owns a 16-row tile (block = 128 rows).
// B tile stream is software-pipelined 2 deep so loads overlap the WMMAs.
// Outputs: qh,kh row-major f16 [BT][64] (post-RoPE), vT f16 [B][64][T].
// ---------------------------------------------------------------------------
__global__ __launch_bounds__(256) void qkv_rope(
    const float* __restrict__ x, const _Float16* __restrict__ wswz,
    _Float16* __restrict__ qh, _Float16* __restrict__ kh,
    _Float16* __restrict__ vT)
{
    const int lane    = threadIdx.x & 31;
    const int wid     = threadIdx.x >> 5;
    const int rowbase = blockIdx.x * 128 + wid * 16;
    const int m       = lane & 15;
    const int kb      = (lane < 16) ? 0 : 16;  // A-matrix per-lane K base
    const int halfsel = lane >> 4;

    const float* xr = x + (size_t)(rowbase + m) * CIN;

    v8f acc[12];                               // [w*4+nt]
#pragma unroll
    for (int t = 0; t < 12; ++t) acc[t] = {};

    // prime the B pipeline (2 tiles in flight)
    v16h B0 = *btile(wswz, 0, lane);
    v16h B1 = *btile(wswz, 1, lane);

    for (int kc = 0; kc < 24; ++kc) {
        const int k0 = kc * 32;
        if (kc + 1 < 24) __builtin_prefetch(xr + k0 + 32, 0, 3); // WGP-scope prefetch

        // A tile (x rows, f32 -> f16), contiguous K order matching B:
        // elems 0..15 = K kb..kb+15; lane<16: kb=0, lane>=16: kb=16
        v4f a0 = *(const v4f*)(xr + k0 + kb);
        v4f a1 = *(const v4f*)(xr + k0 + kb + 4);
        v4f a2 = *(const v4f*)(xr + k0 + kb + 8);
        v4f a3 = *(const v4f*)(xr + k0 + kb + 12);
        v16h A;
#pragma unroll
        for (int i = 0; i < 4; ++i) {
            A[i]      = (_Float16)a0[i];
            A[4 + i]  = (_Float16)a1[i];
            A[8 + i]  = (_Float16)a2[i];
            A[12 + i] = (_Float16)a3[i];
        }

#pragma unroll
        for (int t = 0; t < 12; ++t) {
            v16h Bn = *btile(wswz, kc * 12 + t + 2, lane);  // load 2 ahead
            acc[t]  = WMMA16x16x32(A, B0, acc[t]);
            B0 = B1;
            B1 = Bn;
        }
    }

    // --- RoPE epilogue + stores -------------------------------------------
    // C layout: VGPR r of lane -> row M = r + 8*(lane>>4), col N = lane&15.
    const int trow0 = rowbase + 8 * halfsel;   // rows r=0..7 => t = trow0 + r
#pragma unroll
    for (int nt = 0; nt < 2; ++nt) {
        const int h = nt * 16 + m;             // h in [0,32)
        const float invf = __expf(-((float)(2 * h) / 64.0f) * __logf(10000.0f));
#pragma unroll
        for (int r = 0; r < 8; ++r) {
            const int t     = trow0 + r;
            const int tloc  = t & (T_SEQ - 1); // position within sequence
            const float th  = (float)tloc * invf;
            const float c   = __cosf(th), s = __sinf(th);
            const float qlo = acc[0 + nt][r], qhi = acc[0 + nt + 2][r];
            const float klo = acc[4 + nt][r], khi = acc[4 + nt + 2][r];
            const size_t ro = (size_t)t * HDIM;
            qh[ro + h]      = (_Float16)(qlo * c - qhi * s);
            qh[ro + h + 32] = (_Float16)(qhi * c + qlo * s);
            kh[ro + h]      = (_Float16)(klo * c - khi * s);
            kh[ro + h + 32] = (_Float16)(khi * c + klo * s);
        }
    }
    // V transposed store: vT[b][h][t]; r=0..7 are consecutive t -> one 16B store
    const int bidx  = rowbase >> 8;
    const int tloc0 = (rowbase & (T_SEQ - 1)) + 8 * halfsel;
#pragma unroll
    for (int nt = 0; nt < 4; ++nt) {
        const int h = nt * 16 + m;
        v8h vv;
#pragma unroll
        for (int r = 0; r < 8; ++r) vv[r] = (_Float16)acc[8 + nt][r];
        *(v8h*)(vT + ((size_t)(bidx * HDIM + h)) * T_SEQ + tloc0) = vv;
    }
}

// ---------------------------------------------------------------------------
// Kernel 2: flash attention. Grid (16, 256) = (query tile, batch), 1 wave/blk.
// Online softmax over 32-key chunks; P transposed via 1KB LDS bounce.
// B tiles (K^T and V) are single contiguous 32B loads per lane thanks to the
// contiguous-K B layout and the transposed V buffer.
// ---------------------------------------------------------------------------
__global__ __launch_bounds__(32) void flash_attn(
    const _Float16* __restrict__ qh, const _Float16* __restrict__ kh,
    const _Float16* __restrict__ vT, float* __restrict__ out)
{
    __shared__ __align__(32) _Float16 Pbuf[16 * 32];

    const int lane = threadIdx.x & 31;
    const int m    = lane & 15;
    const int kb   = (lane < 16) ? 0 : 16;    // A-matrix K base (contiguous 16)
    const int kbB  = (lane < 16) ? 0 : 16;    // B-matrix K base (contiguous 16)
    const int half = lane >> 4;
    const int qi   = blockIdx.x;
    const int b    = blockIdx.y;
    const int qbase = qi * 16;

    // Q as two A tiles (head-dim chunks 0..31, 32..63)
    const _Float16* qr = qh + (size_t)(b * T_SEQ + qbase + m) * HDIM;
    v8h q00 = *(const v8h*)(qr + kb);
    v8h q01 = *(const v8h*)(qr + kb + 8);
    v8h q10 = *(const v8h*)(qr + 32 + kb);
    v8h q11 = *(const v8h*)(qr + 32 + kb + 8);
    v16h AQ0 = __builtin_shufflevector(q00, q01, 0,1,2,3,4,5,6,7,8,9,10,11,12,13,14,15);
    v16h AQ1 = __builtin_shufflevector(q10, q11, 0,1,2,3,4,5,6,7,8,9,10,11,12,13,14,15);

    v8f accO[4];
#pragma unroll
    for (int nt = 0; nt < 4; ++nt) accO[nt] = {};
    float mi[8], li[8];
#pragma unroll
    for (int r = 0; r < 8; ++r) { mi[r] = -1e30f; li[r] = 0.0f; }

    const int nchunks = qi / 2 + 1;   // covers keys 0 .. qbase+15
    for (int jc = 0; jc < nchunks; ++jc) {
        const int j0 = jc * 32;

        // S = Q @ K^T for 32 keys (two 16x16 tiles), scale 1/sqrt(64)
        v8f S[2];
#pragma unroll
        for (int s = 0; s < 2; ++s) {
            // B[kk][n] = K[key n][dim kk]; lane column n = m, dims contiguous
            const _Float16* kr = kh + (size_t)(b * T_SEQ + j0 + s * 16 + m) * HDIM;
            v16h BK0 = *(const v16h*)(kr + kbB);        // dims  0..31
            v16h BK1 = *(const v16h*)(kr + 32 + kbB);   // dims 32..63
            v8f z = {};
            z = WMMA16x16x32(AQ0, BK0, z);
            z = WMMA16x16x32(AQ1, BK1, z);
            S[s] = z;
        }

        // online softmax update (rows live in 16-lane halves of C layout)
        float p0[8], p1[8];
#pragma unroll
        for (int r = 0; r < 8; ++r) {
            const int qrow = qbase + r + 8 * half;
            const int key0 = j0 + m;
            const int key1 = j0 + 16 + m;
            float s0 = S[0][r] * 0.125f; if (key0 > qrow) s0 = -1e30f;
            float s1 = S[1][r] * 0.125f; if (key1 > qrow) s1 = -1e30f;
            float mx = fmaxf(s0, s1);
            mx = fmaxf(mx, __shfl_xor(mx, 1, 32));
            mx = fmaxf(mx, __shfl_xor(mx, 2, 32));
            mx = fmaxf(mx, __shfl_xor(mx, 4, 32));
            mx = fmaxf(mx, __shfl_xor(mx, 8, 32));
            const float mn    = fmaxf(mi[r], mx);
            const float alpha = __expf(mi[r] - mn);
            mi[r] = mn;
            p0[r] = __expf(s0 - mn);
            p1[r] = __expf(s1 - mn);
            float rs = p0[r] + p1[r];
            rs += __shfl_xor(rs, 1, 32);
            rs += __shfl_xor(rs, 2, 32);
            rs += __shfl_xor(rs, 4, 32);
            rs += __shfl_xor(rs, 8, 32);
            li[r] = li[r] * alpha + rs;
#pragma unroll
            for (int nt = 0; nt < 4; ++nt) accO[nt][r] *= alpha;
        }

        // transpose P (C layout -> A layout) through LDS
#pragma unroll
        for (int r = 0; r < 8; ++r) {
            const int M = r + 8 * half;
            Pbuf[M * 32 + m]      = (_Float16)p0[r];
            Pbuf[M * 32 + 16 + m] = (_Float16)p1[r];
        }
        asm volatile("s_wait_dscnt 0x0" ::: "memory");
        v8h pa = *(const v8h*)(Pbuf + m * 32 + kb);
        v8h pb = *(const v8h*)(Pbuf + m * 32 + kb + 8);
        asm volatile("s_wait_dscnt 0x0" ::: "memory");
        v16h AP = __builtin_shufflevector(pa, pb, 0,1,2,3,4,5,6,7,8,9,10,11,12,13,14,15);

        // O += P @ V ; V^T layout makes each B-tile one contiguous 32B load
        const _Float16* vb = vT + (size_t)b * HDIM * T_SEQ;
#pragma unroll
        for (int nt = 0; nt < 4; ++nt) {
            const _Float16* vr = vb + (size_t)(nt * 16 + m) * T_SEQ + j0;
            v16h BV = *(const v16h*)(vr + kbB);         // keys j0+kbB .. +15
            accO[nt] = WMMA16x16x32(AP, BV, accO[nt]);
        }
    }

    // final normalize + store fp32 output [1,B,T,H]
#pragma unroll
    for (int nt = 0; nt < 4; ++nt) {
#pragma unroll
        for (int r = 0; r < 8; ++r) {
            const int M = r + 8 * half;
            out[((size_t)(b * T_SEQ + qbase + M)) * HDIM + nt * 16 + m] =
                accO[nt][r] / li[r];
        }
    }
}

// ---------------------------------------------------------------------------
extern "C" void kernel_launch(void* const* d_in, const int* in_sizes, int n_in,
                              void* d_out, int out_size, void* d_ws, size_t ws_size,
                              hipStream_t stream)
{
    (void)in_sizes; (void)n_in; (void)out_size; (void)ws_size;
    const float* x  = (const float*)d_in[0];
    const float* Wq = (const float*)d_in[1];
    const float* Wk = (const float*)d_in[2];
    const float* Wv = (const float*)d_in[3];
    float* out = (float*)d_out;

    // workspace partition (all offsets 256B aligned)
    char* ws = (char*)d_ws;
    _Float16* wswz = (_Float16*)ws;                      // 3*24*4*512*2   = 294,912 B
    _Float16* qhp  = (_Float16*)(ws + 294912);           // 65536*64*2     = 8,388,608 B
    _Float16* khp  = (_Float16*)(ws + 294912 + 8388608);
    _Float16* vTp  = (_Float16*)(ws + 294912 + 2 * 8388608);

    // 1) swizzle weights into WMMA B-register order (f16)
    swizzle_weights<<<(3 * 24 * 4 * 512 + 255) / 256, 256, 0, stream>>>(Wq, Wk, Wv, wswz);

    // 2) fused QKV projection + RoPE: 65536 rows / 128 rows per block
    qkv_rope<<<BT / 128, 256, 0, stream>>>(x, wswz, qhp, khp, vTp);

    // 3) flash attention: (query tile, batch)
    flash_attn<<<dim3(T_SEQ / 16, 256), 32, 0, stream>>>(qhp, khp, vTp, out);
}
